// CachedMLP_47614007444156
// MI455X (gfx1250) — matrix operand order repaired and checked by reading, not compiled, not run on hardware
//
#include <hip/hip_runtime.h>
#include <hip/hip_bf16.h>
#include <cstdint>
#include <cstddef>

// ---------------------------------------------------------------------------
// MoE 2-expert MLP for MI455X (gfx1250, wave32, WMMA).
//   T=128 tokens, D=4096, ACTIVE=11468 rows per weight matrix.
//   Weights = 1.13 GB f32 -> HBM-bound (~48us @ 23.3 TB/s). bf16 WMMA
//   (f32 accum) keeps compute well below the memory floor; f32 WMMA (K=4)
//   would be compute-bound. Double-buffered LDS + register prefetch keeps
//   HBM streaming overlapped with the WMMAs. f32->bf16x2 packing is done
//   with one v_perm_b32 (+2 adds for rounding) to keep VALU off the
//   critical path.
// ---------------------------------------------------------------------------

#define TOKENS 128
#define D_IN   4096
#define ACTIVE 11468
#define A_PAD  11520   // 180 tiles of 64, zero padded

typedef __attribute__((ext_vector_type(16))) __bf16 v16bf;
typedef __attribute__((ext_vector_type(8)))  float  v8f;

struct Q2 { uint4 lo, hi; };

static __device__ __forceinline__ v16bf make_frag(uint4 lo, uint4 hi) {
    Q2 q{lo, hi};
    return __builtin_bit_cast(v16bf, q);
}

// round-half-up bf16 (1 add + shift); ties differ from RNE by 1 ulp only
static __device__ __forceinline__ uint32_t f32_to_bf16bits(float f) {
    return (__builtin_bit_cast(uint32_t, f) + 0x8000u) >> 16;
}

// pack two f32 -> {bf16(hi), bf16(lo)} : 2x v_add_u32 + 1x v_perm_b32
static __device__ __forceinline__ uint32_t pack_bf16x2(float lo, float hi) {
    const uint32_t a = __builtin_bit_cast(uint32_t, lo) + 0x8000u;
    const uint32_t b = __builtin_bit_cast(uint32_t, hi) + 0x8000u;
    // dst = { b[31:24], b[23:16], a[31:24], a[23:16] }
    return __builtin_amdgcn_perm(b, a, 0x07060302u);
}

// ---------------------------------------------------------------------------
// Phase 1: G[e][t][a] = ew[e] * silu(x@w1_e^T) * (x@w3_e^T), stored bf16.
// Block tile: M=128 tokens x N=64 active rows. 8 waves = 4(M) x 2(N),
// each wave owns a 32x32 region = 2x2 WMMA tiles per matrix.
// Pipeline: regs(k+1) global loads issued while WMMAs consume LDS[k&1].
// ---------------------------------------------------------------------------
struct PreG {
    float4 x[4];
    float4 w1[2];
    float4 w3[2];
};

static __device__ __forceinline__ PreG p1_load(const float* __restrict__ x,
                                               const float* __restrict__ w1,
                                               const float* __restrict__ w3,
                                               int tid, int tileA, int k0) {
    PreG r;
    const int xr = tid >> 3;          // base row within 32-row slab
    const int c4 = tid & 7;           // float4 column
    #pragma unroll
    for (int i = 0; i < 4; ++i)
        r.x[i] = ((const float4*)(x + (size_t)(i * 32 + xr) * D_IN + k0))[c4];
    #pragma unroll
    for (int i = 0; i < 2; ++i) {
        const int ag = tileA + i * 32 + xr;
        float4 v1 = make_float4(0.f, 0.f, 0.f, 0.f);
        float4 v3 = make_float4(0.f, 0.f, 0.f, 0.f);
        if (ag < ACTIVE) {
            v1 = ((const float4*)(w1 + (size_t)ag * D_IN + k0))[c4];
            v3 = ((const float4*)(w3 + (size_t)ag * D_IN + k0))[c4];
        }
        r.w1[i] = v1;
        r.w3[i] = v3;
    }
    return r;
}

__global__ __launch_bounds__(256)
void moe_gateup_kernel(const float* __restrict__ x,
                       const float* __restrict__ w1_e0,
                       const float* __restrict__ w3_e0,
                       const float* __restrict__ w1_e1,
                       const float* __restrict__ w3_e1,
                       const float* __restrict__ ewt,
                       const int*   __restrict__ eids,
                       uint16_t*    __restrict__ Gbase)
{
    const int e = blockIdx.y;
    const float* __restrict__ w1 = e ? w1_e1 : w1_e0;
    const float* __restrict__ w3 = e ? w3_e1 : w3_e0;
    const int swap = (eids[0] != 0) ? 1 : 0;
    const float ew = ewt[swap ? (1 - e) : e];
    uint16_t* __restrict__ G = Gbase + (size_t)e * TOKENS * A_PAD;

    const int tileA = blockIdx.x * 64;

    __shared__ uint32_t sX [2][128 * 16];   // 128 x 32 bf16 (2 per dword)
    __shared__ uint32_t sW1[2][ 64 * 16];
    __shared__ uint32_t sW3[2][ 64 * 16];

    const int tid  = threadIdx.x;
    const int lane = tid & 31;
    const int wave = tid >> 5;
    const int wm   = wave & 3;           // token block: rows 32*wm
    const int wn   = wave >> 2;          // A block:    cols 32*wn
    const int lrow = lane & 15;
    const int lhi  = lane >> 4;          // 0/1: lane half

    v8f acc1[2][2], acc3[2][2];
    #pragma unroll
    for (int i = 0; i < 2; ++i)
        #pragma unroll
        for (int j = 0; j < 2; ++j) { acc1[i][j] = (v8f)0.0f; acc3[i][j] = (v8f)0.0f; }

    const int xr = tid >> 3;
    const int c4 = tid & 7;

    PreG pre = p1_load(x, w1, w3, tid, tileA, 0);

    for (int k0 = 0; k0 < D_IN; k0 += 32) {
        const int buf = (k0 >> 5) & 1;

        // ---- stage prefetched chunk into LDS[buf] (bf16 packed dwords) ----
        #pragma unroll
        for (int i = 0; i < 4; ++i) {
            const float4 v = pre.x[i];
            uint32_t* p = &sX[buf][(i * 32 + xr) * 16 + c4 * 2];
            p[0] = pack_bf16x2(v.x, v.y);
            p[1] = pack_bf16x2(v.z, v.w);
        }
        #pragma unroll
        for (int i = 0; i < 2; ++i) {
            const float4 v1 = pre.w1[i];
            const float4 v3 = pre.w3[i];
            uint32_t* p1 = &sW1[buf][(i * 32 + xr) * 16 + c4 * 2];
            uint32_t* p3 = &sW3[buf][(i * 32 + xr) * 16 + c4 * 2];
            p1[0] = pack_bf16x2(v1.x, v1.y);
            p1[1] = pack_bf16x2(v1.z, v1.w);
            p3[0] = pack_bf16x2(v3.x, v3.y);
            p3[1] = pack_bf16x2(v3.z, v3.w);
        }
        __syncthreads();   // single barrier/iter: ping-pong makes it safe

        // ---- issue next chunk's global loads (overlap with WMMAs) ----
        if (k0 + 32 < D_IN)
            pre = p1_load(x, w1, w3, tid, tileA, k0 + 32);

        // ---- A fragments (16x32 bf16): lane half picks K base 0/16; chunks
        //      [kb, kb+8) and [kb+8, kb+16) (contiguous, matches B) ----
        const int kb_u = lhi * 8;                // dword offset of low chunk
        v16bf afrag[2];
        #pragma unroll
        for (int tm = 0; tm < 2; ++tm) {
            const uint32_t* p = &sX[buf][(32 * wm + 16 * tm + lrow) * 16];
            afrag[tm] = make_frag(*(const uint4*)(p + kb_u),
                                  *(const uint4*)(p + kb_u + 4));
        }
        // ---- B fragments (32x16): lane = column, contiguous 16 K values ----
        const int bk_u = lhi * 8;
        v16bf b1f[2], b3f[2];
        #pragma unroll
        for (int tn = 0; tn < 2; ++tn) {
            const int r = 32 * wn + 16 * tn + lrow;
            const uint32_t* p1 = &sW1[buf][r * 16];
            const uint32_t* p3 = &sW3[buf][r * 16];
            b1f[tn] = make_frag(*(const uint4*)(p1 + bk_u), *(const uint4*)(p1 + bk_u + 4));
            b3f[tn] = make_frag(*(const uint4*)(p3 + bk_u), *(const uint4*)(p3 + bk_u + 4));
        }
        #pragma unroll
        for (int tm = 0; tm < 2; ++tm)
            #pragma unroll
            for (int tn = 0; tn < 2; ++tn) {
                acc1[tm][tn] = __builtin_amdgcn_wmma_f32_16x16x32_bf16(
                    false, afrag[tm], false, b1f[tn], (short)0, acc1[tm][tn], false, false);
                acc3[tm][tn] = __builtin_amdgcn_wmma_f32_16x16x32_bf16(
                    false, afrag[tm], false, b3f[tn], (short)0, acc3[tm][tn], false, false);
            }
    }

    // ---- epilogue: silu(gate) * up * ew -> bf16 G[t][a] ----
    #pragma unroll
    for (int tm = 0; tm < 2; ++tm)
        #pragma unroll
        for (int tn = 0; tn < 2; ++tn) {
            const int row0 = 32 * wm + 16 * tm + 8 * lhi;   // token
            const int col  = tileA + 32 * wn + 16 * tn + lrow;
            #pragma unroll
            for (int v = 0; v < 8; ++v) {
                const float g = acc1[tm][tn][v];
                const float u = acc3[tm][tn][v];
                const float s = g / (1.0f + __expf(-g));    // silu
                const float val = ew * s * u;
                G[(size_t)(row0 + v) * A_PAD + col] = (uint16_t)f32_to_bf16bits(val);
            }
        }
}

// ---------------------------------------------------------------------------
// Phase 2: out[t][d] = sum_e sum_a G[e][t][a] * w2_e[a][d]   (ew folded in G)
// Block tile: M=128 tokens x N=64 of D. Flattened K loop over both experts.
// w2 staged transposed in LDS as packed bf16x2 dwords (each thread loads two
// consecutive a-rows so the k-pair packs into one u32 -> ds_store_b32).
// G fragments loaded straight from workspace (bf16, L2-resident).
// ---------------------------------------------------------------------------
struct PreW2 {
    float4 lo;   // row a0 + 2*ap
    float4 hi;   // row a0 + 2*ap + 1
};

static __device__ __forceinline__ PreW2 p2_load(const float* __restrict__ w2,
                                                int tid, int tileD, int a0) {
    PreW2 r;
    const int c4 = tid & 15;          // float4 column along d (coalesced)
    const int ap = tid >> 4;          // a-pair index 0..15
    float4 vl = make_float4(0.f, 0.f, 0.f, 0.f);
    float4 vh = make_float4(0.f, 0.f, 0.f, 0.f);
    const int ag = a0 + 2 * ap;
    if (ag < ACTIVE)
        vl = ((const float4*)(w2 + (size_t)ag * D_IN + tileD))[c4];
    if (ag + 1 < ACTIVE)
        vh = ((const float4*)(w2 + (size_t)(ag + 1) * D_IN + tileD))[c4];
    r.lo = vl;
    r.hi = vh;
    return r;
}

__global__ __launch_bounds__(256)
void moe_down_kernel(const uint16_t* __restrict__ G,
                     const float*    __restrict__ w2_e0,
                     const float*    __restrict__ w2_e1,
                     float*          __restrict__ out)
{
    const int tileD = blockIdx.x * 64;

    __shared__ uint32_t sW2[2][64 * 16];   // [d in tile][k-pair], bf16x2

    const int tid  = threadIdx.x;
    const int lane = tid & 31;
    const int wave = tid >> 5;
    const int wm   = wave & 3;
    const int wn   = wave >> 2;
    const int lrow = lane & 15;
    const int lhi  = lane >> 4;

    v8f acc[2][2];
    #pragma unroll
    for (int i = 0; i < 2; ++i)
        #pragma unroll
        for (int j = 0; j < 2; ++j) acc[i][j] = (v8f)0.0f;

    const int NK  = A_PAD / 32;   // 360 chunks per expert
    const int TOT = 2 * NK;       // both experts, flattened

    const int c4 = tid & 15;

    const int ap = tid >> 4;

    PreW2 pre = p2_load(w2_e0, tid, tileD, 0);

    for (int it = 0; it < TOT; ++it) {
        const int buf = it & 1;
        const int ecur = (it >= NK) ? 1 : 0;
        const int a0 = (it - ecur * NK) * 32;

        // ---- stage prefetched w2 chunk transposed: 4x ds_store_b32 ----
        {
            const float4 vl = pre.lo;
            const float4 vh = pre.hi;
            const int d = c4 * 4;
            sW2[buf][(d + 0) * 16 + ap] = pack_bf16x2(vl.x, vh.x);
            sW2[buf][(d + 1) * 16 + ap] = pack_bf16x2(vl.y, vh.y);
            sW2[buf][(d + 2) * 16 + ap] = pack_bf16x2(vl.z, vh.z);
            sW2[buf][(d + 3) * 16 + ap] = pack_bf16x2(vl.w, vh.w);
        }
        __syncthreads();   // single barrier/iter: ping-pong makes it safe

        // ---- issue next chunk's global loads (overlap with WMMAs) ----
        if (it + 1 < TOT) {
            const int nit = it + 1;
            const int en = (nit >= NK) ? 1 : 0;
            const float* w2n = en ? w2_e1 : w2_e0;
            pre = p2_load(w2n, tid, tileD, (nit - en * NK) * 32);
        }

        // ---- A fragments: G rows (tokens), direct global bf16 loads ----
        const uint16_t* __restrict__ Ge = G + (size_t)ecur * TOKENS * A_PAD;
        const int kb = lhi * 16;             // bf16 offset of low chunk
        v16bf afrag[2];
        #pragma unroll
        for (int tm = 0; tm < 2; ++tm) {
            const int r = 32 * wm + 16 * tm + lrow;
            const uint16_t* gp = Ge + (size_t)r * A_PAD + a0;
            afrag[tm] = make_frag(*(const uint4*)(gp + kb),
                                  *(const uint4*)(gp + kb + 8));
        }
        // ---- B fragments from transposed LDS: contiguous 16 K ----
        v16bf bfrag[2];
        #pragma unroll
        for (int tn = 0; tn < 2; ++tn) {
            const int d = 32 * wn + 16 * tn + lrow;
            const uint32_t* p = &sW2[buf][d * 16 + lhi * 8];
            bfrag[tn] = make_frag(*(const uint4*)p, *(const uint4*)(p + 4));
        }
        #pragma unroll
        for (int tm = 0; tm < 2; ++tm)
            #pragma unroll
            for (int tn = 0; tn < 2; ++tn)
                acc[tm][tn] = __builtin_amdgcn_wmma_f32_16x16x32_bf16(
                    false, afrag[tm], false, bfrag[tn], (short)0, acc[tm][tn], false, false);
    }

    // ---- store f32 output ----
    #pragma unroll
    for (int tm = 0; tm < 2; ++tm)
        #pragma unroll
        for (int tn = 0; tn < 2; ++tn) {
            const int row0 = 32 * wm + 16 * tm + 8 * lhi;
            const int col  = tileD + 32 * wn + 16 * tn + lrow;
            #pragma unroll
            for (int v = 0; v < 8; ++v)
                out[(size_t)(row0 + v) * D_IN + col] = acc[tm][tn][v];
        }
}

// ---------------------------------------------------------------------------
extern "C" void kernel_launch(void* const* d_in, const int* in_sizes, int n_in,
                              void* d_out, int out_size, void* d_ws, size_t ws_size,
                              hipStream_t stream) {
    const float* x     = (const float*)d_in[0];
    const float* ewt   = (const float*)d_in[1];
    const int*   eids  = (const int*)  d_in[2];
    const float* w1_e0 = (const float*)d_in[3];
    const float* w3_e0 = (const float*)d_in[4];
    const float* w2_e0 = (const float*)d_in[5];
    const float* w1_e1 = (const float*)d_in[6];
    const float* w3_e1 = (const float*)d_in[7];
    const float* w2_e1 = (const float*)d_in[8];

    uint16_t* G  = (uint16_t*)d_ws;          // 2 x 128 x 11520 bf16 ~= 5.6 MB
    float*   out = (float*)d_out;

    dim3 g1(A_PAD / 64, 2);
    moe_gateup_kernel<<<g1, 256, 0, stream>>>(x, w1_e0, w3_e0, w1_e1, w3_e1,
                                              ewt, eids, G);
    moe_down_kernel<<<dim3(D_IN / 64), 256, 0, stream>>>(G, w2_e0, w2_e1, out);
}
